// ScaleDotProductAttention_3607772528847
// MI455X (gfx1250) — matrix operand order repaired and checked by reading, not compiled, not run on hardware
//
#include <hip/hip_runtime.h>
#include <math.h>

typedef __attribute__((ext_vector_type(16))) _Float16 v16h;
typedef __attribute__((ext_vector_type(8)))  float    v8f;
typedef __attribute__((ext_vector_type(4)))  float    v4f;

#define ATT_B 2
#define ATT_H 16
#define ATT_L 2048
#define ATT_D 64
#define NKT   (ATT_L / 16)       // 128 k-tiles
#define WAVES 8
#define SROW  (ATT_L + 4)        // padded LDS row stride: 2052 % 64 == 4 -> conflict-free
#define SCALE 0.125f             // 1/sqrt(64)
#define VPACK_ELEMS ((size_t)ATT_B * ATT_H * ATT_L * ATT_D)

// ---- WMMA operand packing in this code's contiguous k order (k = 16*half + s) ----
// A (16x32 f16): lane<16 -> row M=lane, K=0..15 contiguous; lane>=16 -> K=16..31.
__device__ __forceinline__ v16h pack_a_f32(const float* rowp, int s1, int s2, float sc) {
    v16h r;
#pragma unroll
    for (int i = 0; i < 8; ++i) r[i] = (_Float16)(rowp[s1 + i] * sc);
#pragma unroll
    for (int i = 0; i < 8; ++i) r[8 + i] = (_Float16)(rowp[s2 + i] * sc);
    return r;
}
// B (32x16 f16): lane<16 -> col N=lane, K=0..15 contiguous; lane>=16 -> K=16..31.
__device__ __forceinline__ v16h pack_b_contig(const float* rowp, int s) {
    v16h r;
#pragma unroll
    for (int i = 0; i < 16; ++i) r[i] = (_Float16)rowp[s + i];
    return r;
}

// Pre-pack V (f32, [bh][k][d]) into f16 B-operand lane layout:
// dst[((bh*64 + s)*64 + col)*32 + half*16 + i] = V[bh][s*32 + half*16 + i][col]
// so a Phase-D lane reads its 16 K-values as one aligned 32-byte block.
__global__ __launch_bounds__(256) void pack_v_kernel(const float* __restrict__ v,
                                                     _Float16* __restrict__ vp) {
    const int tid  = blockIdx.x * 256 + threadIdx.x;   // flat over [bh][k][col]
    const int col  = tid & 63;
    const int kk   = (tid >> 6) & (ATT_L - 1);
    const int bh   = tid >> 17;
    const int s    = kk >> 5;
    const int half = (kk >> 4) & 1;
    const int i    = kk & 15;
    vp[((((size_t)bh * (ATT_L / 32) + s) * ATT_D + col) << 5) + half * 16 + i] =
        (_Float16)v[tid];
}

__global__ __launch_bounds__(256) void
attention_fused_kernel(const float* __restrict__ q,
                       const float* __restrict__ k,
                       const float* __restrict__ v,
                       const _Float16* __restrict__ vpack,
                       float* __restrict__ out,
                       float* __restrict__ score,
                       int packed) {
    extern __shared__ float smem[];
    float* sS    = smem;                    // [16][SROW] raw scores -> exp values
    float* sPart = smem + 16 * SROW;        // [4][256] PV partial tiles
    float* red   = sPart + 1024;            // [16][16] reduction scratch
    float* rowM  = red + 256;               // [16] row max
    float* rowI  = rowM + 16;               // [16] 1/rowsum

    const int tid  = threadIdx.x;
    const int lane = tid & 31;
    const int wv   = tid >> 5;              // wave 0..7
    const int half = lane >> 4;             // 0/1
    const int l16  = lane & 15;

    const int bh = blockIdx.x >> 7;         // b*H + h
    const int qt = blockIdx.x & 127;        // 16-row q tile

    const float* qb = q + (size_t)bh * ATT_L * ATT_D;
    const float* kb = k + (size_t)bh * ATT_L * ATT_D;
    const float* vb = v + (size_t)bh * ATT_L * ATT_D;

    // ---------------- Phase A: S = (Q*scale) K^T ----------------
    const float* qrow = qb + (size_t)(qt * 16 + l16) * ATT_D;
    const v16h aQ0 = pack_a_f32(qrow, 0  + 16 * half, 8 + 16 * half, SCALE);
    const v16h aQ1 = pack_a_f32(qrow, 32 + 16 * half, 40 + 16 * half, SCALE);

    for (int it = 0; it < NKT / WAVES; ++it) {
        const int kt = wv * (NKT / WAVES) + it;
        const float* krow = kb + (size_t)(kt * 16 + l16) * ATT_D;
        const v16h b0 = pack_b_contig(krow, 0  + 16 * half);
        const v16h b1 = pack_b_contig(krow, 32 + 16 * half);

        v8f acc = {};
        acc = __builtin_amdgcn_wmma_f32_16x16x32_f16(false, aQ0, false, b0,
                                                     (short)0, acc, false, false);
        acc = __builtin_amdgcn_wmma_f32_16x16x32_f16(false, aQ1, false, b1,
                                                     (short)0, acc, false, false);
        // C/D layout: elem r -> row m = r + 8*half, col n = l16
#pragma unroll
        for (int r = 0; r < 8; ++r)
            sS[(r + 8 * half) * SROW + kt * 16 + l16] = acc[r];
    }
    __syncthreads();

    // ---------------- Phase B: row max, then cache exp and sum ----------------
    const int row = tid & 15;
    const int seg = tid >> 4;               // 16 segments of 128 columns
    float* sp = sS + row * SROW + seg * 128;
    {
        float m = -1e30f;
#pragma unroll 8
        for (int i = 0; i < 32; ++i) {
            const v4f x = *(const v4f*)(sp + 4 * i);
            m = fmaxf(fmaxf(fmaxf(m, x.x), fmaxf(x.y, x.z)), x.w);
        }
        red[seg * 16 + row] = m;
    }
    __syncthreads();
    if (tid < 16) {
        float m = red[tid];
#pragma unroll
        for (int s = 1; s < 16; ++s) m = fmaxf(m, red[s * 16 + tid]);
        rowM[tid] = m;
    }
    __syncthreads();
    const float rm = rowM[row];
    {
        float sum = 0.f;
#pragma unroll 4
        for (int i = 0; i < 32; ++i) {
            v4f x = *(const v4f*)(sp + 4 * i);
            x.x = __expf(x.x - rm); x.y = __expf(x.y - rm);
            x.z = __expf(x.z - rm); x.w = __expf(x.w - rm);
            *(v4f*)(sp + 4 * i) = x;           // cache exp(s - max) in LDS
            sum += (x.x + x.y) + (x.z + x.w);
        }
        red[seg * 16 + row] = sum;
    }
    __syncthreads();
    if (tid < 16) {
        float s = 0.f;
#pragma unroll
        for (int t = 0; t < 16; ++t) s += red[t * 16 + tid];
        rowI[tid] = 1.f / s;
    }
    __syncthreads();

    // ---------------- Phase C: NT-stream normalized score to global ----------------
    // (no LDS write-back: Phase D folds 1/rowsum into the A-operand conversion)
    {
        const float inv = rowI[row];
        v4f* gs = (v4f*)(score + ((size_t)bh * ATT_L + qt * 16 + row) * ATT_L + seg * 128);
#pragma unroll 4
        for (int i = 0; i < 32; ++i) {
            v4f x = *(const v4f*)(sp + 4 * i);
            x *= inv;
            __builtin_nontemporal_store(x, gs + i);  // 537MB stream: don't pollute L2
        }
    }
    __syncthreads();

    // ---------------- Phase D: out = (P/rowsum) V (split-K across wave pairs) ----------------
    const int colTile = wv & 3;             // 16-col slice of D
    const int khalf   = wv >> 2;            // K range [khalf*1024, +1024)
    const int col     = 16 * colTile + l16;
    const float invRow = rowI[l16];         // A-operand row is l16 for every lane
    const float* prow = sS + (size_t)l16 * SROW + khalf * 1024;
    const float* vhb  = vb + (size_t)khalf * 1024 * ATT_D;
    const _Float16* vpb = vpack
        + ((((size_t)bh * (ATT_L / 32) + khalf * 32) * ATT_D + col) << 5) + half * 16;

    v8f acc = {};
    for (int s = 0; s < 32; ++s) {
        const int k0 = s * 32;
        const v16h a = pack_a_f32(prow, k0 + 16 * half, k0 + 8 + 16 * half, invRow);
        v16h b;
        if (packed) {
            // one aligned 32B block per lane: 2x global_load_b128, zero converts
            b = *(const v16h*)(vpb + ((size_t)s * ATT_D << 5));
        } else {
            const float* vp = vhb + (size_t)(k0 + 16 * half) * ATT_D + col;
#pragma unroll
            for (int i = 0; i < 16; ++i) b[i] = (_Float16)vp[(size_t)i * ATT_D];
        }
        acc = __builtin_amdgcn_wmma_f32_16x16x32_f16(false, a, false, b,
                                                     (short)0, acc, false, false);
    }
    __syncthreads();
    if (khalf == 1) {
#pragma unroll
        for (int r = 0; r < 8; ++r)
            sPart[colTile * 256 + (r + 8 * half) * 16 + l16] = acc[r];
    }
    __syncthreads();
    if (khalf == 0) {
#pragma unroll
        for (int r = 0; r < 8; ++r) {
            const int m = r + 8 * half;
            out[((size_t)bh * ATT_L + qt * 16 + m) * ATT_D + col] =
                acc[r] + sPart[colTile * 256 + m * 16 + l16];
        }
    }
}

extern "C" void kernel_launch(void* const* d_in, const int* in_sizes, int n_in,
                              void* d_out, int out_size, void* d_ws, size_t ws_size,
                              hipStream_t stream) {
    const float* q = (const float*)d_in[0];
    const float* k = (const float*)d_in[1];
    const float* v = (const float*)d_in[2];
    float* out   = (float*)d_out;                                  // [B,H,L,D]
    float* score = out + (size_t)ATT_B * ATT_H * ATT_L * ATT_D;    // [B,H,L,L]

    _Float16* vpack = (_Float16*)d_ws;
    const int packed = (ws_size >= VPACK_ELEMS * sizeof(_Float16)) ? 1 : 0;
    if (packed) {
        const int pblocks = (int)(VPACK_ELEMS / 256);              // 16384
        pack_v_kernel<<<pblocks, 256, 0, stream>>>(v, vpack);
    }

    const int blocks = ATT_B * ATT_H * (ATT_L / 16);               // 4096
    const size_t lds = (size_t)(16 * SROW + 1024 + 256 + 16 + 16) * sizeof(float);
    attention_fused_kernel<<<blocks, 256, lds, stream>>>(q, k, v, vpack, out, score, packed);
}
